// GAT_22153441312996
// MI455X (gfx1250) — matrix operand order repaired and checked by reading, not compiled, not run on hardware
//
#include <hip/hip_runtime.h>
#include <hip/hip_bf16.h>

// ---------------------------------------------------------------------------
// Types for CDNA5 WMMA (wave32): 16x16x32 bf16 -> f32
// ---------------------------------------------------------------------------
typedef __attribute__((ext_vector_type(16))) __bf16 v16bf;
typedef __attribute__((ext_vector_type(8)))  float  v8f;
typedef __attribute__((ext_vector_type(4)))  float  v4f;

__device__ __forceinline__ unsigned short f2bf_bits(float f) {
    // round-to-nearest-even f32 -> bf16 (bit-level)
    unsigned u = __float_as_uint(f);
    return (unsigned short)((u + 0x7FFFu + ((u >> 16) & 1u)) >> 16);
}

__device__ __forceinline__ __bf16 f2bf(float f) {
    unsigned short us = f2bf_bits(f);
    __bf16 b;
    __builtin_memcpy(&b, &us, 2);
    return b;
}

__device__ __forceinline__ float lrelu(float x) { return x > 0.f ? x : 0.2f * x; }

// Monotone float <-> ordered-uint mapping for atomicMax on f32 values
__device__ __forceinline__ unsigned fkey(float f) {
    unsigned b = __float_as_uint(f);
    return (b & 0x80000000u) ? ~b : (b | 0x80000000u);
}
__device__ __forceinline__ float fdekey(unsigned u) {
    unsigned b = (u & 0x80000000u) ? (u & 0x7FFFFFFFu) : ~u;
    return __uint_as_float(b);
}

// ---------------------------------------------------------------------------
// Pack W[K, Nout] (f32, row-major) into per-lane WMMA B-fragment order (bf16).
// Element (kt, t, lane, j) = bf16( W[(kt*32 + khalf(lane) + j) * Nout + t*16 + lane%16] )
// Column masking (Nout < NT*16) resolved here with zeros.
// ---------------------------------------------------------------------------
__global__ void gat_pack_w(const float* __restrict__ W, unsigned short* __restrict__ Wp,
                           int NT, int KT, int Nout)
{
    int idx   = blockIdx.x * blockDim.x + threadIdx.x;
    int total = KT * NT * 512;
    if (idx >= total) return;
    int j    = idx & 15;
    int lane = (idx >> 4) & 31;
    int t    = (idx >> 9) % NT;
    int kt   = idx / (NT * 512);
    int khalf = (lane & 16) ? 16 : 0;
    int n     = t * 16 + (lane & 15);
    int k     = kt * 32 + khalf + j;
    float v   = (n < Nout) ? W[(size_t)k * Nout + n] : 0.f;
    Wp[idx]   = f2bf_bits(v);
}

// ---------------------------------------------------------------------------
// GEMM: Hout[N, Nout] = X[N, K] * W[K, Nout],  K = KT*32, Nout <= NT*16.
// One wave per 16-row M tile (8 tiles per 256-thread block).
// Packed bf16 W is staged in LDS once per block; B fragments come from
// ds_load_b128 pairs, A fragments from unconditional global_load_b128s.
// ---------------------------------------------------------------------------
template<int NT, int KT>
__global__ __launch_bounds__(256)
void gat_gemm_wmma(const float* __restrict__ X, const unsigned short* __restrict__ Wp,
                   float* __restrict__ Hout, int N, int Nout)
{
    const int K = KT * 32;
    __shared__ __align__(32) unsigned short smem[KT * NT * 512];

    // ---- cooperative copy of packed W into LDS (vectorized) ----
    {
        const int nVec = (KT * NT * 512) / 8;            // uint4 chunks (16B)
        const uint4* src = (const uint4*)Wp;
        uint4*       dst = (uint4*)smem;
        for (int i = threadIdx.x; i < nVec; i += 256) dst[i] = src[i];
    }
    __syncthreads();

    const int wave  = threadIdx.x >> 5;
    const int lane  = threadIdx.x & 31;
    const int tile  = blockIdx.x * 8 + wave;
    const int mtile = (N + 15) >> 4;

    if (tile < mtile) {                       // wave-uniform guard (after barrier)
        const int mbase  = tile << 4;
        const int mrow   = mbase + (lane & 15);
        const int mrowc  = (mrow < N) ? mrow : (N - 1);   // clamp; garbage rows never stored
        const int khalfA = (lane & 16) ? 16 : 0;          // A: lanes 16-31 hold K+16..K+31
        const int rowoff = (lane & 16) ? 8 : 0;
        const int ncol   = lane & 15;
        const float* xrow = X + (size_t)mrowc * K + khalfA;

        v8f acc[NT] = {};

        #pragma unroll
        for (int kt = 0; kt < KT; ++kt) {
            // ---- A fragment: 4x global_load_b128 + in-register bf16 convert ----
            const float* xr = xrow + kt * 32;
            v4f x0 = *(const v4f*)(xr + 0);
            v4f x1 = *(const v4f*)(xr + 4);
            v4f x2 = *(const v4f*)(xr + 8);
            v4f x3 = *(const v4f*)(xr + 12);
            v16bf a;
            #pragma unroll
            for (int j = 0; j < 4; ++j) {
                a[j]      = f2bf(x0[j]);
                a[j + 4]  = f2bf(x1[j]);
                a[j + 8]  = f2bf(x2[j]);
                a[j + 12] = f2bf(x3[j]);
            }

            #pragma unroll
            for (int t = 0; t < NT; ++t) {
                // ---- B fragment straight from LDS (pre-packed layout) ----
                v16bf b = *(const v16bf*)(smem + ((kt * NT + t) * 32 + lane) * 16);
                acc[t] = __builtin_amdgcn_wmma_f32_16x16x32_bf16(
                             false, a, false, b, (short)0, acc[t], false, false);
            }
        }

        // ---- store D: element r -> row mbase + r + rowoff, col t*16 + ncol ----
        #pragma unroll
        for (int t = 0; t < NT; ++t) {
            const int col = t * 16 + ncol;
            const bool cv = (col < Nout);
            #pragma unroll
            for (int r = 0; r < 8; ++r) {
                const int row = mbase + r + rowoff;
                if (cv && row < N) Hout[(size_t)row * Nout + col] = acc[t][r];
            }
        }
    }
}

// ---------------------------------------------------------------------------
// Per-node attention logits: a_s[n,h] = <h[n,h,:], att_src[h,:]>, same for a_d
// ---------------------------------------------------------------------------
template<int H, int C>
__global__ void gat_node_attn(const float* __restrict__ Hf,
                              const float* __restrict__ att_s,
                              const float* __restrict__ att_d,
                              float* __restrict__ a_s, float* __restrict__ a_d, int N)
{
    int idx = blockIdx.x * blockDim.x + threadIdx.x;
    if (idx >= N * H) return;
    int n = idx / H, h = idx % H;
    const float* row = Hf + (size_t)n * (H * C) + h * C;
    float s = 0.f, d = 0.f;
    #pragma unroll
    for (int c = 0; c < C; ++c) {
        s += row[c] * att_s[h * C + c];
        d += row[c] * att_d[h * C + c];
    }
    a_s[idx] = s;
    a_d[idx] = d;
}

// ---------------------------------------------------------------------------
// Edge pass 1: segment max over destinations (ordered-uint atomicMax)
// Edges e >= E are the synthesized self-loops (i,i).
// ---------------------------------------------------------------------------
template<int H>
__global__ void gat_edge_max(const int* __restrict__ eidx, int E, int N,
                             const float* __restrict__ a_s, const float* __restrict__ a_d,
                             unsigned* __restrict__ mkey)
{
    int idx = blockIdx.x * blockDim.x + threadIdx.x;
    int Et  = E + N;
    if (idx >= Et * H) return;
    int e = idx / H, h = idx % H;
    int s, d;
    if (e < E) { s = eidx[e]; d = eidx[E + e]; } else { s = d = e - E; }
    float v = lrelu(a_s[s * H + h] + a_d[d * H + h]);
    atomicMax(&mkey[d * H + h], fkey(v));
}

// ---------------------------------------------------------------------------
// Edge pass 2: segment sum of exp(score - max)
// ---------------------------------------------------------------------------
template<int H>
__global__ void gat_edge_expsum(const int* __restrict__ eidx, int E, int N,
                                const float* __restrict__ a_s, const float* __restrict__ a_d,
                                const unsigned* __restrict__ mkey, float* __restrict__ ssum)
{
    int idx = blockIdx.x * blockDim.x + threadIdx.x;
    int Et  = E + N;
    if (idx >= Et * H) return;
    int e = idx / H, h = idx % H;
    int s, d;
    if (e < E) { s = eidx[e]; d = eidx[E + e]; } else { s = d = e - E; }
    float v  = lrelu(a_s[s * H + h] + a_d[d * H + h]);
    float ee = __expf(v - fdekey(mkey[d * H + h]));
    atomicAdd(&ssum[d * H + h], ee);
}

// ---------------------------------------------------------------------------
// Edge pass 3: out[dst] += h[src] * alpha   (64 threads per edge, f32 atomics)
// ---------------------------------------------------------------------------
template<int H, int C>
__global__ __launch_bounds__(256)
void gat_edge_aggregate(const int* __restrict__ eidx, int E, int N,
                        const float* __restrict__ Hf,
                        const float* __restrict__ a_s, const float* __restrict__ a_d,
                        const unsigned* __restrict__ mkey, const float* __restrict__ ssum,
                        float* __restrict__ out)
{
    const int Wd = H * C;                     // 64 or 40
    int e = blockIdx.x * 4 + (threadIdx.x >> 6);
    int c = threadIdx.x & 63;
    int Et = E + N;
    if (e >= Et || c >= Wd) return;
    int s, d;
    if (e < E) { s = eidx[e]; d = eidx[E + e]; } else { s = d = e - E; }
    int h = c / C;
    float v     = lrelu(a_s[s * H + h] + a_d[d * H + h]);
    float ee    = __expf(v - fdekey(mkey[d * H + h]));
    float alpha = ee / (ssum[d * H + h] + 1e-16f);
    atomicAdd(&out[(size_t)d * Wd + c], Hf[(size_t)s * Wd + c] * alpha);
}

// ---------------------------------------------------------------------------
// Utility kernels
// ---------------------------------------------------------------------------
__global__ void gat_fill_u32(unsigned* __restrict__ p, unsigned v, int n)
{
    int i = blockIdx.x * blockDim.x + threadIdx.x;
    if (i < n) p[i] = v;
}

__global__ void gat_bias_add64(float* __restrict__ out, const float* __restrict__ b, int N)
{
    int i = blockIdx.x * blockDim.x + threadIdx.x;
    if (i >= N * 64) return;
    out[i] += b[i & 63];
}

__global__ void gat_bias_logsoftmax(float* __restrict__ out, const float* __restrict__ b,
                                    int N, int NC)
{
    int n = blockIdx.x * blockDim.x + threadIdx.x;
    if (n >= N) return;
    float* row = out + (size_t)n * NC;
    float m = -3.0e38f;
    for (int c = 0; c < NC; ++c) {
        float x = row[c] + b[c];
        row[c] = x;
        if (x > m) m = x;
    }
    float sum = 0.f;
    for (int c = 0; c < NC; ++c) sum += __expf(row[c] - m);
    float lse = m + __logf(sum);
    for (int c = 0; c < NC; ++c) row[c] -= lse;
}

// ---------------------------------------------------------------------------
// Launcher
// ---------------------------------------------------------------------------
extern "C" void kernel_launch(void* const* d_in, const int* in_sizes, int n_in,
                              void* d_out, int out_size, void* d_ws, size_t ws_size,
                              hipStream_t stream)
{
    const float* feat = (const float*)d_in[0];
    const int*   eidx = (const int*)d_in[1];
    const float* W0   = (const float*)d_in[2];
    const float* as0  = (const float*)d_in[3];
    const float* ad0  = (const float*)d_in[4];
    const float* b0   = (const float*)d_in[5];
    const float* W1   = (const float*)d_in[6];
    const float* as1  = (const float*)d_in[7];
    const float* ad1  = (const float*)d_in[8];
    const float* b1   = (const float*)d_in[9];
    const float* W2   = (const float*)d_in[10];
    const float* as2  = (const float*)d_in[11];
    const float* ad2  = (const float*)d_in[12];
    const float* b2   = (const float*)d_in[13];
    float* out = (float*)d_out;

    const int FIN = in_sizes[2] / 64;        // 256
    const int N   = in_sizes[0] / FIN;       // 100000
    const int E   = in_sizes[1] / 2;         // 3200000
    const int NC  = in_sizes[10] / 64;       // 40
    const int Et  = E + N;
    (void)n_in; (void)out_size; (void)ws_size; (void)FIN;

    // ---- carve workspace (all 256B aligned) ----
    auto al = [](size_t x) { return (x + 255) & ~(size_t)255; };
    char* ws = (char*)d_ws;
    size_t off = 0;
    float*          bufH = (float*)         (ws + off); off += al((size_t)N * 64 * 4);
    float*          bufO = (float*)         (ws + off); off += al((size_t)N * 64 * 4);
    float*          aS   = (float*)         (ws + off); off += al((size_t)N * 8 * 4);
    float*          aD   = (float*)         (ws + off); off += al((size_t)N * 8 * 4);
    unsigned*       mkey = (unsigned*)      (ws + off); off += al((size_t)N * 8 * 4);
    float*          ssum = (float*)         (ws + off); off += al((size_t)N * 8 * 4);
    unsigned short* Wp   = (unsigned short*)(ws + off); off += al((size_t)8 * 4 * 512 * 2);

    const int mtiles   = (N + 15) / 16;
    const int gemmGrid = (mtiles + 7) / 8;
    const int naGrid8  = (N * 8 + 255) / 256;
    const int naGrid1  = (N + 255) / 256;
    const int edGrid8  = (Et * 8 + 255) / 256;
    const int edGrid1  = (Et + 255) / 256;
    const int agGrid   = (Et + 3) / 4;

    auto fill0 = [&](void* p, int nWords) {
        gat_fill_u32<<<(nWords + 255) / 256, 256, 0, stream>>>((unsigned*)p, 0u, nWords);
    };
    auto packW = [&](const float* W, int NT, int KT, int Nout) {
        int total = KT * NT * 512;
        gat_pack_w<<<(total + 255) / 256, 256, 0, stream>>>(W, Wp, NT, KT, Nout);
    };

    // =================== Layer 0: 256 -> 8x8 (concat 64) ===================
    packW(W0, 4, 8, 64);
    gat_gemm_wmma<4, 8><<<gemmGrid, 256, 0, stream>>>(feat, Wp, bufH, N, 64);
    gat_node_attn<8, 8><<<naGrid8, 256, 0, stream>>>(bufH, as0, ad0, aS, aD, N);
    fill0(mkey, N * 8); fill0(ssum, N * 8); fill0(bufO, N * 64);
    gat_edge_max<8><<<edGrid8, 256, 0, stream>>>(eidx, E, N, aS, aD, mkey);
    gat_edge_expsum<8><<<edGrid8, 256, 0, stream>>>(eidx, E, N, aS, aD, mkey, ssum);
    gat_edge_aggregate<8, 8><<<agGrid, 256, 0, stream>>>(eidx, E, N, bufH, aS, aD, mkey, ssum, bufO);
    gat_bias_add64<<<(N * 64 + 255) / 256, 256, 0, stream>>>(bufO, b0, N);

    // =================== Layer 1: 64 -> 8x8 (concat 64) ====================
    packW(W1, 4, 2, 64);
    gat_gemm_wmma<4, 2><<<gemmGrid, 256, 0, stream>>>(bufO, Wp, bufH, N, 64);
    gat_node_attn<8, 8><<<naGrid8, 256, 0, stream>>>(bufH, as1, ad1, aS, aD, N);
    fill0(mkey, N * 8); fill0(ssum, N * 8);
    gat_edge_max<8><<<edGrid8, 256, 0, stream>>>(eidx, E, N, aS, aD, mkey);
    gat_edge_expsum<8><<<edGrid8, 256, 0, stream>>>(eidx, E, N, aS, aD, mkey, ssum);
    fill0(bufO, N * 64);        // reuse: GEMM above already consumed old bufO
    gat_edge_aggregate<8, 8><<<agGrid, 256, 0, stream>>>(eidx, E, N, bufH, aS, aD, mkey, ssum, bufO);
    gat_bias_add64<<<(N * 64 + 255) / 256, 256, 0, stream>>>(bufO, b1, N);

    // =================== Layer 2: 64 -> 1x40 (mean == identity) ============
    packW(W2, 3, 2, NC);
    gat_gemm_wmma<3, 2><<<gemmGrid, 256, 0, stream>>>(bufO, Wp, bufH, N, NC);
    gat_node_attn<1, 40><<<naGrid1, 256, 0, stream>>>(bufH, as2, ad2, aS, aD, N);
    fill0(mkey, N); fill0(ssum, N); fill0(out, N * NC);
    gat_edge_max<1><<<edGrid1, 256, 0, stream>>>(eidx, E, N, aS, aD, mkey);
    gat_edge_expsum<1><<<edGrid1, 256, 0, stream>>>(eidx, E, N, aS, aD, mkey, ssum);
    gat_edge_aggregate<1, 40><<<agGrid, 256, 0, stream>>>(eidx, E, N, bufH, aS, aD, mkey, ssum, out);
    gat_bias_logsoftmax<<<naGrid1, 256, 0, stream>>>(out, b2, N, NC);
}
